// GCNNet_24524263260957
// MI455X (gfx1250) — matrix operand order repaired and checked by reading, not compiled, not run on hardware
//
#include <hip/hip_runtime.h>

#define DIM 20

typedef __attribute__((ext_vector_type(2)))  float    v2f;
typedef __attribute__((ext_vector_type(8)))  float    v8f;
typedef __attribute__((ext_vector_type(16))) _Float16 v16h;

#if __has_builtin(__builtin_amdgcn_wmma_f32_16x16x4_f32)
#define USE_WMMA_F32_4 1
#endif

// ---------------- degree / normalization ----------------
__global__ void k_init_deg(float* __restrict__ deg, int n) {
    int i = blockIdx.x * blockDim.x + threadIdx.x;
    if (i < n) deg[i] = 1.0f;  // self-loop contributes 1
}

__global__ void k_deg_scatter(const int* __restrict__ dst, float* __restrict__ deg, int e) {
    int i = blockIdx.x * blockDim.x + threadIdx.x;
    if (i < e) atomicAdd(&deg[dst[i]], 1.0f);
}

__global__ void k_dinv(const float* __restrict__ deg, float* __restrict__ dinv, int n) {
    int i = blockIdx.x * blockDim.x + threadIdx.x;
    if (i < n) {
        float d = deg[i];
        dinv[i] = (d > 0.0f) ? rsqrtf(d) : 0.0f;  // d >= 1 always, but mirror reference
    }
}

// ---------------- dense h @ W via WMMA ----------------
// Each wave computes a 16x20 output tile: C[16x20] = H[16x20] @ W[20x20].
// W is zero-padded to 32x32 in LDS. f32 path: 5 K-chunks of 4 x 2 N-tiles
// = 10 v_wmma_f32_16x16x4_f32 per wave. EXEC stays all-ones for WMMA
// (wave-uniform early exit only; per-lane guards are on stores, after WMMA).
__global__ __launch_bounds__(256) void k_gemm20(const float* __restrict__ H,
                                                const int*   __restrict__ gather,
                                                const float* __restrict__ W,
                                                float* __restrict__ out, int n) {
    __shared__ float Ws[32 * 32];
    for (int i = threadIdx.x; i < 32 * 32; i += blockDim.x) {
        int r = i >> 5, c = i & 31;
        Ws[i] = (r < DIM && c < DIM) ? W[r * DIM + c] : 0.0f;
    }
    __syncthreads();

    int wave = threadIdx.x >> 5;   // wave32
    int lane = threadIdx.x & 31;
    int rowBase = (blockIdx.x * 8 + wave) * 16;
    if (rowBase >= n) return;      // wave-uniform

    int half = lane >> 4;          // 0: lanes 0-15, 1: lanes 16-31
    int l16  = lane & 15;
    int row  = rowBase + l16;
    int rsafe = row < n ? row : (n - 1);                 // clamp for loads only
    int src   = gather ? gather[rsafe] : rsafe;
    const float* hrow = H + (long)src * DIM;

    v8f c0 = {}; v8f c1 = {};

#ifdef USE_WMMA_F32_4
    // A 16x4 f32: lanes 0-15 hold K=k0+0,k0+1 ; lanes 16-31 hold K=k0+2,k0+3
    int koff = half * 2;
    #pragma unroll
    for (int k = 0; k < 5; ++k) {
        int k0 = k * 4;
        v2f a;  a.x  = hrow[k0 + koff];            a.y  = hrow[k0 + koff + 1];
        v2f b0; b0.x = Ws[(k0 + koff) * 32 + l16]; b0.y = Ws[(k0 + koff + 1) * 32 + l16];
        v2f b1; b1.x = Ws[(k0 + koff) * 32 + 16 + l16];
                b1.y = Ws[(k0 + koff + 1) * 32 + 16 + l16];
        c0 = __builtin_amdgcn_wmma_f32_16x16x4_f32(false, a, false, b0, (short)0, c0, false, false);
        c1 = __builtin_amdgcn_wmma_f32_16x16x4_f32(false, a, false, b1, (short)0, c1, false, false);
    }
#else
    // Fallback: f16 inputs, f32 accumulate, K padded 20->32.
    // A 16x32 f16 layout: elem h of v16h at lane L -> K = (h/8)*16 + (L<16?0:8) + h%8
    // B 32x16 f16 layout: elem h at lane L -> K = (h/8)*16 + (L<16?0:8) + h%8, N = L%16
    v16h a, b0, b1;
    #pragma unroll
    for (int h = 0; h < 16; ++h) {
        int Ka = (h >> 3) * 16 + half * 8 + (h & 7);
        a[h] = (Ka < DIM) ? (_Float16)hrow[Ka] : (_Float16)0.0f;
        int Kb = (h >> 3) * 16 + half * 8 + (h & 7);
        b0[h] = (_Float16)Ws[Kb * 32 + l16];        // zero-padded already
        b1[h] = (_Float16)Ws[Kb * 32 + 16 + l16];
    }
    c0 = __builtin_amdgcn_wmma_f32_16x16x32_f16(false, a, false, b0, (short)0, c0, false, false);
    c1 = __builtin_amdgcn_wmma_f32_16x16x32_f16(false, a, false, b1, (short)0, c1, false, false);
#endif

    // C/D layout: VGPR j holds M=j (lanes 0-15) / M=j+8 (lanes 16-31), N=lane%16
    int mbase = half * 8;
    #pragma unroll
    for (int j = 0; j < 8; ++j) {
        int r = rowBase + mbase + j;
        if (r < n) {
            out[(long)r * DIM + l16] = c0[j];
            if (l16 < DIM - 16) out[(long)r * DIM + 16 + l16] = c1[j];
        }
    }
}

// ---------------- aggregation ----------------
// agg[i][c] = dinv[i]^2 * hw[i][c] + b[c]   (self-loop term + bias)
__global__ void k_init_agg(const float* __restrict__ hw, const float* __restrict__ dinv,
                           const float* __restrict__ b, float* __restrict__ agg, int n) {
    int i = blockIdx.x * blockDim.x + threadIdx.x;
    if (i < n * DIM) {
        int node = i / DIM, c = i % DIM;
        float di = dinv[node];
        agg[i] = di * di * hw[i] + b[c];
    }
}

// agg[dst] += dinv[src]*dinv[dst] * hw[src]  per edge (f32 atomics, L2-resident)
__global__ void k_edge_scatter(const int* __restrict__ src, const int* __restrict__ dst,
                               const float* __restrict__ dinv, const float* __restrict__ hw,
                               float* __restrict__ agg, int e) {
    int i = blockIdx.x * blockDim.x + threadIdx.x;
    if (i >= e) return;
    int s = src[i], d = dst[i];
    float w = dinv[s] * dinv[d];
    const float4* hp = (const float4*)(hw + (long)s * DIM);  // 80B rows, 16B aligned
    float* ap = agg + (long)d * DIM;
    #pragma unroll
    for (int q = 0; q < 5; ++q) {
        float4 v = hp[q];
        atomicAdd(ap + q * 4 + 0, w * v.x);
        atomicAdd(ap + q * 4 + 1, w * v.y);
        atomicAdd(ap + q * 4 + 2, w * v.z);
        atomicAdd(ap + q * 4 + 3, w * v.w);
    }
}

__global__ void k_relu(const float* __restrict__ in, float* __restrict__ out, int n) {
    int i = blockIdx.x * blockDim.x + threadIdx.x;
    if (i < n) out[i] = fmaxf(in[i], 0.0f);
}

// ---------------- driver ----------------
extern "C" void kernel_launch(void* const* d_in, const int* in_sizes, int n_in,
                              void* d_out, int out_size, void* d_ws, size_t ws_size,
                              hipStream_t stream) {
    const int*   x    = (const int*)d_in[0];    // [N] node ids
    const int*   edge = (const int*)d_in[1];    // [2,E] int32
    const float* emb  = (const float*)d_in[2];  // [N,20]
    const float* W1   = (const float*)d_in[3];  // [20,20]
    const float* b1   = (const float*)d_in[4];  // [20]
    const float* W2   = (const float*)d_in[5];
    const float* b2   = (const float*)d_in[6];
    int n = in_sizes[0];
    int e = in_sizes[1] / 2;
    const int* srcp = edge;
    const int* dstp = edge + e;
    float* out = (float*)d_out;

    // workspace carve-up (256B aligned slices)
    char* ws = (char*)d_ws;
    auto take = [&](size_t bytes) {
        char* p = ws;
        ws += (bytes + 255) & ~(size_t)255;
        return p;
    };
    float* deg  = (float*)take((size_t)n * sizeof(float));
    float* dinv = (float*)take((size_t)n * sizeof(float));
    float* hw   = (float*)take((size_t)n * DIM * sizeof(float));  // h @ W scratch
    float* agg  = (float*)take((size_t)n * DIM * sizeof(float));  // layer-1 aggregate
    float* h1   = (float*)take((size_t)n * DIM * sizeof(float));  // relu output

    const int B = 256;
    int gN  = (n + B - 1) / B;
    int gE  = (e + B - 1) / B;
    int gND = ((long)n * DIM + B - 1) / B;
    int rowTiles = (n + 15) / 16;
    int gG  = (rowTiles + 7) / 8;   // 8 waves per block, 16 rows per wave

    // normalization: deg = 1 + in-degree ; dinv = deg^-1/2
    k_init_deg   <<<gN, B, 0, stream>>>(deg, n);
    k_deg_scatter<<<gE, B, 0, stream>>>(dstp, deg, e);
    k_dinv       <<<gN, B, 0, stream>>>(deg, dinv, n);

    // layer 1: hw = emb[x] @ W1 ; agg = dinv^2*hw + b1 ; scatter ; h1 = relu(agg)
    k_gemm20      <<<gG,  B, 0, stream>>>(emb, x, W1, hw, n);
    k_init_agg    <<<gND, B, 0, stream>>>(hw, dinv, b1, agg, n);
    k_edge_scatter<<<gE,  B, 0, stream>>>(srcp, dstp, dinv, hw, agg, e);
    k_relu        <<<gND, B, 0, stream>>>(agg, h1, (int)((long)n * DIM));

    // layer 2: hw = h1 @ W2 ; out = dinv^2*hw + b2 ; scatter into d_out
    k_gemm20      <<<gG,  B, 0, stream>>>(h1, nullptr, W2, hw, n);
    k_init_agg    <<<gND, B, 0, stream>>>(hw, dinv, b2, out, n);
    k_edge_scatter<<<gE,  B, 0, stream>>>(srcp, dstp, dinv, hw, out, e);
}
